// GAM_76373108458042
// MI455X (gfx1250) — matrix operand order repaired and checked by reading, not compiled, not run on hardware
//
#include <hip/hip_runtime.h>

typedef __attribute__((ext_vector_type(16))) _Float16 v16h;
typedef __attribute__((ext_vector_type(8)))  _Float16 v8h;
typedef __attribute__((ext_vector_type(8)))  float    v8f;
typedef __attribute__((ext_vector_type(4)))  int      v4i;

#define HW 4096
#define CH 256
#define NB 8
#define LOG2E 1.4426950408889634f
#define LPAD 264   /* padded LDS row stride in f16 elements (528 B) */

#if __has_builtin(__builtin_amdgcn_global_load_async_to_lds_b128)
#define HAVE_ASYNC_LDS 1
#else
#define HAVE_ASYNC_LDS 0
#endif

typedef __attribute__((address_space(1))) v4i gv4i_t;   // global int4
typedef __attribute__((address_space(3))) v4i lv4i_t;   // LDS int4

// copy 16 bytes global -> LDS (async DMA if available, else through VGPRs)
__device__ __forceinline__ void cp16B(const void* g, void* l) {
#if HAVE_ASYNC_LDS
  __builtin_amdgcn_global_load_async_to_lds_b128(
      (gv4i_t*)(unsigned long long)g,
      (lv4i_t*)(unsigned)(unsigned long long)l, 0, 0);
#else
  *(uint4*)l = *(const uint4*)g;
#endif
}

__device__ __forceinline__ void wait_async() {
#if HAVE_ASYNC_LDS
#if __has_builtin(__builtin_amdgcn_s_wait_asynccnt)
  __builtin_amdgcn_s_wait_asynccnt(0);
#else
  asm volatile("s_wait_asynccnt 0x0" ::: "memory");
#endif
#endif
}

// ---------------------------------------------------------------------------
// K1: 1x1-conv projection  Y[b,n,o] = sum_c W[o,c]*x[b,n,c] + bias[o]
// writes Yt as f16 [b][n][c] (channel-contiguous) + per-n channel mean/max.
// ---------------------------------------------------------------------------
__global__ __launch_bounds__(256)
void proj_kernel(const float* __restrict__ x, const float* __restrict__ w,
                 const float* __restrict__ bias, _Float16* __restrict__ Yt,
                 float* __restrict__ qmean, float* __restrict__ qmax)
{
  __shared__ float xs[16][CH];
  __shared__ float redS[8];
  __shared__ float redM[8];
  const int t    = threadIdx.x;
  const int lane = t & 31;
  const int wv   = t >> 5;
  const int b    = blockIdx.x >> 8;
  const int n0   = (blockIdx.x & 255) << 4;

  const float* xb = x + ((size_t)b * HW + n0) * CH;
  #pragma unroll
  for (int i = 0; i < 16; ++i) xs[i][t] = xb[(size_t)i * CH + t];
  __syncthreads();

  float acc[16];
  const float bv = bias[t];
  #pragma unroll
  for (int i = 0; i < 16; ++i) acc[i] = bv;

  const float4* wr = (const float4*)(w + (size_t)t * CH);
  for (int c4 = 0; c4 < CH / 4; ++c4) {
    const float4 wvv = wr[c4];
    const int c = c4 * 4;
    #pragma unroll
    for (int i = 0; i < 16; ++i) {
      acc[i] = fmaf(wvv.x, xs[i][c + 0], acc[i]);
      acc[i] = fmaf(wvv.y, xs[i][c + 1], acc[i]);
      acc[i] = fmaf(wvv.z, xs[i][c + 2], acc[i]);
      acc[i] = fmaf(wvv.w, xs[i][c + 3], acc[i]);
    }
  }

  _Float16* yb = Yt + ((size_t)b * HW + n0) * CH;
  #pragma unroll
  for (int i = 0; i < 16; ++i) yb[(size_t)i * CH + t] = (_Float16)acc[i];

  for (int i = 0; i < 16; ++i) {
    float s = acc[i], m = acc[i];
    #pragma unroll
    for (int d = 16; d > 0; d >>= 1) {
      s += __shfl_xor(s, d);
      m = fmaxf(m, __shfl_xor(m, d));
    }
    if (lane == 0) { redS[wv] = s; redM[wv] = m; }
    __syncthreads();
    if (t == 0) {
      float S = 0.f, M = -3.0e38f;
      #pragma unroll
      for (int j = 0; j < 8; ++j) { S += redS[j]; M = fmaxf(M, redM[j]); }
      qmean[(size_t)b * HW + n0 + i] = S * (1.0f / CH);
      qmax [(size_t)b * HW + n0 + i] = M;
    }
    __syncthreads();
  }
}

// ---------------------------------------------------------------------------
// Pass 1: per-row softmax stats of S = Yt * Yt^T (flash style, base-2 domain).
// Wave owns a 16-row tile (A hoisted to 64 VGPRs); the streamed B tile
// (shared by all 8 waves) is double-buffered in LDS via async copies.
// All 8 B chunks are pulled into registers before the 8-WMMA accum chain so
// only one dscnt wait fences the whole chain.
// Writes rowpack[m] = (M2, q_mean/Z, q_max/Z, 0).
// ---------------------------------------------------------------------------
__global__ __launch_bounds__(256)
void rowstats_kernel(const _Float16* __restrict__ Yt,
                     const float* __restrict__ qmean, const float* __restrict__ qmax,
                     float4* __restrict__ rowpack)
{
  __shared__ _Float16 btile[2][16][LPAD];   // 2 x 8.25 KB
  const int t    = threadIdx.x;
  const int lane = t & 31;
  const int wv   = t >> 5;
  const int b    = blockIdx.x >> 5;
  const int mb   = blockIdx.x & 31;
  const int m0   = mb * 128 + wv * 16;
  const int g    = lane >> 4;
  const int r    = lane & 15;
  const _Float16* yb = Yt + (size_t)b * HW * CH;

  // staging role: thread copies 32 B of row (t>>4), elem chunk (t&15)*16
  const int srow = t >> 4;
  const int scol = (t & 15) * 16;

  // hoist A tiles (16 rows x K=256) into registers, contiguous k = 16*g + s per lane half
  v16h a_t[8];
  const _Float16* arow = yb + (size_t)(m0 + r) * CH;
  #pragma unroll
  for (int k = 0; k < 8; ++k) {
    const int c0 = k * 32;
    a_t[k] = *(const v16h*)(arow + c0 + 16 * g);
  }

  // prologue: stage tile 0
  {
    const _Float16* gsrc = yb + (size_t)srow * CH + scol;
    cp16B(gsrc,     &btile[0][srow][scol]);
    cp16B(gsrc + 8, &btile[0][srow][scol + 8]);
  }
  wait_async();
  __syncthreads();

  float Mp[8], Zp[8];
  #pragma unroll
  for (int j = 0; j < 8; ++j) { Mp[j] = -3.0e38f; Zp[j] = 0.f; }

  const int NT = HW / 16;
  for (int tile = 0; tile < NT; ++tile) {
    const int nxt = tile + 1;
    if (nxt < NT) {                     // prefetch next tile into other buffer
      const _Float16* gsrc = yb + ((size_t)nxt * 16 + srow) * CH + scol;
      _Float16* ldst = &btile[nxt & 1][srow][scol];
      cp16B(gsrc,     ldst);
      cp16B(gsrc + 8, ldst + 8);
    }

    // pull all 8 B chunks into registers, then run the WMMA accum chain
    const _Float16* lb = &btile[tile & 1][r][16 * g];
    v16h bt[8];
    #pragma unroll
    for (int k = 0; k < 8; ++k) bt[k] = *(const v16h*)(lb + 32 * k);

    v8f acc = {};
    #pragma unroll
    for (int k = 0; k < 8; ++k)
      acc = __builtin_amdgcn_wmma_f32_16x16x32_f16(false, a_t[k], false, bt[k],
                                                   (short)0, acc, false, false);

    // online softmax stats in exp2 domain; rescale branch is rare
    #pragma unroll
    for (int j = 0; j < 8; ++j) {
      const float s2 = acc[j] * LOG2E;
      if (s2 > Mp[j]) {
        Zp[j] *= __builtin_amdgcn_exp2f(Mp[j] - s2);
        Mp[j] = s2;
      }
      Zp[j] += __builtin_amdgcn_exp2f(s2 - Mp[j]);
    }

    wait_async();
    __syncthreads();
  }

  // merge (M,Z) across the 16 lanes of each half, write packed row stats
  #pragma unroll
  for (int j = 0; j < 8; ++j) {
    float M = Mp[j], Z = Zp[j];
    #pragma unroll
    for (int d = 1; d < 16; d <<= 1) {
      const float Mo = __shfl_xor(M, d);
      const float Zo = __shfl_xor(Z, d);
      const float Mn = fmaxf(M, Mo);
      Z = Z * __builtin_amdgcn_exp2f(M - Mn) + Zo * __builtin_amdgcn_exp2f(Mo - Mn);
      M = Mn;
    }
    if (r == 0) {
      const size_t gi = (size_t)b * HW + m0 + j + 8 * g;
      const float iz = 1.0f / Z;
      rowpack[gi] = make_float4(M, qmean[gi] * iz, qmax[gi] * iz, 0.f);
    }
  }
}

// ---------------------------------------------------------------------------
// Pass 2: f1[n] = sum_m A1[m]*exp2(S2[m,n]-M2[m]) (f3 with A3), fused:
// fmap[b,n] = w0*f1 + w1*f3 + b0.  Wave owns 16 columns (B hoisted); streamed
// A tile + its rowpack are double-buffered in LDS via async copies.
// ---------------------------------------------------------------------------
__global__ __launch_bounds__(256)
void attnpool_kernel(const _Float16* __restrict__ Yt,
                     const float4* __restrict__ rowpack,
                     const float* __restrict__ fw, const float* __restrict__ fb,
                     float* __restrict__ fmap)
{
  __shared__ _Float16 atile[2][16][LPAD];
  __shared__ float4 rtile[2][16];
  const int t    = threadIdx.x;
  const int lane = t & 31;
  const int wv   = t >> 5;
  const int b    = blockIdx.x >> 5;
  const int nb   = blockIdx.x & 31;
  const int nw   = nb * 128 + wv * 16;
  const int g    = lane >> 4;
  const int r    = lane & 15;
  const _Float16* yb = Yt + (size_t)b * HW * CH;
  const float4* rpb  = rowpack + (size_t)b * HW;

  const int srow = t >> 4;
  const int scol = (t & 15) * 16;

  // hoist B tiles for our 16 columns
  v16h b_t[8];
  const _Float16* brow = yb + (size_t)(nw + r) * CH + 16 * g;
  #pragma unroll
  for (int k = 0; k < 8; ++k) b_t[k] = *(const v16h*)(brow + k * 32);

  // prologue: stage m-tile 0 (+ its row stats)
  {
    const _Float16* gsrc = yb + (size_t)srow * CH + scol;
    cp16B(gsrc,     &atile[0][srow][scol]);
    cp16B(gsrc + 8, &atile[0][srow][scol + 8]);
    if (t < 16) cp16B(&rpb[t], &rtile[0][t]);
  }
  wait_async();
  __syncthreads();

  float f1 = 0.f, f3 = 0.f;
  const int NT = HW / 16;
  for (int tile = 0; tile < NT; ++tile) {
    const int nxt = tile + 1;
    if (nxt < NT) {
      const _Float16* gsrc = yb + ((size_t)nxt * 16 + srow) * CH + scol;
      _Float16* ldst = &atile[nxt & 1][srow][scol];
      cp16B(gsrc,     ldst);
      cp16B(gsrc + 8, ldst + 8);
      if (t < 16) cp16B(&rpb[nxt * 16 + t], &rtile[nxt & 1][t]);
    }

    // pull all 8 A chunks into registers, then run the WMMA accum chain
    const _Float16* la = &atile[tile & 1][r][0];
    v16h at[8];
    #pragma unroll
    for (int k = 0; k < 8; ++k) {
      const int c0 = k * 32;
      at[k] = *(const v16h*)(la + c0 + 16 * g);
    }

    v8f acc = {};
    #pragma unroll
    for (int k = 0; k < 8; ++k)
      acc = __builtin_amdgcn_wmma_f32_16x16x32_f16(false, at[k], false, b_t[k],
                                                   (short)0, acc, false, false);

    #pragma unroll
    for (int j = 0; j < 8; ++j) {
      const float4 rp = rtile[tile & 1][j + 8 * g];
      const float e = __builtin_amdgcn_exp2f(fmaf(acc[j], LOG2E, -rp.x));
      f1 = fmaf(rp.y, e, f1);
      f3 = fmaf(rp.z, e, f3);
    }

    wait_async();
    __syncthreads();
  }

  // lanes L and L+16 hold the same column n = nw + r over disjoint m subsets
  f1 += __shfl_xor(f1, 16);
  f3 += __shfl_xor(f3, 16);
  if (g == 0) {
    const float w0 = fw[0], w1 = fw[1], b0 = fb[0];
    fmap[(size_t)b * HW + nw + r] = w0 * f1 + w1 * f3 + b0;
  }
}

// ---------------------------------------------------------------------------
// K4: out[b,n,c] = x[b,n,c] * fmap[b,n]   (float4 vectorized)
// ---------------------------------------------------------------------------
__global__ __launch_bounds__(256)
void gate_kernel(const float* __restrict__ x, const float* __restrict__ fmap,
                 float* __restrict__ out)
{
  const size_t idx  = (size_t)blockIdx.x * 256 + threadIdx.x;
  const size_t base = idx * 4;
  const size_t n    = base >> 8;
  const float fm = fmap[n];
  const float4 xv = *(const float4*)(x + base);
  float4 ov;
  ov.x = xv.x * fm; ov.y = xv.y * fm; ov.z = xv.z * fm; ov.w = xv.w * fm;
  *(float4*)(out + base) = ov;
}

// ---------------------------------------------------------------------------
extern "C" void kernel_launch(void* const* d_in, const int* in_sizes, int n_in,
                              void* d_out, int out_size, void* d_ws, size_t ws_size,
                              hipStream_t stream)
{
  const float* x    = (const float*)d_in[0];
  const float* w    = (const float*)d_in[1];
  const float* bias = (const float*)d_in[2];
  const float* fw   = (const float*)d_in[3];
  const float* fb   = (const float*)d_in[4];
  float* out = (float*)d_out;

  char* ws = (char*)d_ws;
  const size_t nBH = (size_t)NB * HW;
  _Float16* Yt = (_Float16*)ws;
  size_t off = nBH * CH * sizeof(_Float16);          // 16.78 MB
  float4* rowpack = (float4*)(ws + off); off += nBH * sizeof(float4);
  float* qmean    = (float*)(ws + off);  off += nBH * sizeof(float);
  float* qmax     = (float*)(ws + off);  off += nBH * sizeof(float);
  float* fmap     = (float*)(ws + off);  off += nBH * sizeof(float);

  proj_kernel    <<<NB * HW / 16, 256, 0, stream>>>(x, w, bias, Yt, qmean, qmax);
  rowstats_kernel<<<NB * 32,      256, 0, stream>>>(Yt, qmean, qmax, rowpack);
  attnpool_kernel<<<NB * 32,      256, 0, stream>>>(Yt, rowpack, fw, fb, fmap);
  gate_kernel    <<<(unsigned)((nBH * CH / 4) / 256), 256, 0, stream>>>(x, fmap, out);
}
